// RWKVSeq_81277961109610
// MI455X (gfx1250) — matrix operand order repaired and checked by reading, not compiled, not run on hardware
//
#include <hip/hip_runtime.h>
#include <hip/hip_bf16.h>

// ---------------------------------------------------------------------------
// RWKV spatial-mix layer for MI455X (gfx1250, wave32, WMMA bf16 tensor cores,
// async global->LDS tile staging).  B=8, N=4096 (16x16x16), C=512
// ---------------------------------------------------------------------------

#define Bx 8
#define Nx 4096
#define Cx 512
#define Mx (Bx * Nx)          // 32768 rows
#define NCH 16                // WKV chunks
#define CT (Nx / NCH)         // 256 steps per chunk

typedef __attribute__((ext_vector_type(16))) __bf16 v16bf;
typedef __attribute__((ext_vector_type(8)))  float  v8f;

__device__ __forceinline__ unsigned short f2bf(float f) {
    unsigned int u = __float_as_uint(f);
    unsigned int r = (u + 0x7FFFu + ((u >> 16) & 1u)) >> 16;   // RNE
    return (unsigned short)r;
}

// Async DMA: 16 bytes per lane, global -> LDS, tracked by ASYNCcnt.
__device__ __forceinline__ void async_b128(unsigned lds_off, const unsigned short* g) {
    asm volatile("global_load_async_to_lds_b128 %0, %1, off"
                 :: "v"(lds_off), "v"(g)
                 : "memory");
}
__device__ __forceinline__ void wait_async0() {
    asm volatile("s_wait_asynccnt 0" ::: "memory");
}

// ---------------------------------------------------------------------------
// Kernel 0: convert the four 512x512 fp32 weight matrices to bf16
// ---------------------------------------------------------------------------
__global__ void convert_weights_kernel(const float* __restrict__ wk, const float* __restrict__ wv,
                                       const float* __restrict__ wr, const float* __restrict__ wo,
                                       unsigned short* __restrict__ ok, unsigned short* __restrict__ ov,
                                       unsigned short* __restrict__ orr, unsigned short* __restrict__ oo) {
    int t = blockIdx.x * 256 + threadIdx.x;       // 0 .. 512*512-1
    ok[t]  = f2bf(wk[t]);
    ov[t]  = f2bf(wv[t]);
    orr[t] = f2bf(wr[t]);
    oo[t]  = f2bf(wo[t]);
}

// ---------------------------------------------------------------------------
// Kernel 1: 3D q_shift + token mixing -> xk, xv, xr in bf16
// ---------------------------------------------------------------------------
__global__ void qshift_mix_kernel(const float* __restrict__ tok,
                                  const float* __restrict__ mk, const float* __restrict__ mv,
                                  const float* __restrict__ mr,
                                  unsigned short* __restrict__ xk, unsigned short* __restrict__ xv,
                                  unsigned short* __restrict__ xr) {
    size_t t = (size_t)blockIdx.x * 256 + threadIdx.x;   // B*N*C threads
    int c  = (int)(t & (Cx - 1));
    size_t bn = t >> 9;                                  // /512
    int n  = (int)(bn & (Nx - 1));
    int b  = (int)(bn >> 12);                            // /4096
    int w_ = n & 15, h = (n >> 4) & 15, d = n >> 8;

    int grp = c / 85;                 // g = C//6 = 85; c in [510,512) -> 6
    if (grp > 6) grp = 6;
    int  src = n;
    bool ok  = true;
    switch (grp) {
        case 0: ok = (w_ >= 1);  src = n - 1;   break;   // +W
        case 1: ok = (w_ <= 14); src = n + 1;   break;   // -W
        case 2: ok = (h  >= 1);  src = n - 16;  break;   // +H
        case 3: ok = (h  <= 14); src = n + 16;  break;   // -H
        case 4: ok = (d  >= 1);  src = n - 256; break;   // +D
        case 5: ok = (d  <= 14); src = n + 256; break;   // -D
        default: break;                                   // unshifted tail
    }
    float x  = tok[t];
    float xx = ok ? tok[(((size_t)b * Nx + src) << 9) + c] : 0.0f;

    float k_ = mk[c], v_ = mv[c], r_ = mr[c];
    xk[t] = f2bf(x * k_ + xx * (1.0f - k_));
    xv[t] = f2bf(x * v_ + xx * (1.0f - v_));
    xr[t] = f2bf(x * r_ + xx * (1.0f - r_));
}

// ---------------------------------------------------------------------------
// Kernel 2: bf16 WMMA GEMM  C[M,512] = A[M,512] * W[512,512]^T  (K contiguous
// on both sides). BM=128, BN=64, BK=64, 8 waves, 32x32 tile per wave.
// Double-buffered LDS, async global->LDS staging (ASYNCcnt).
// act: 0 = none, 1 = sigmoid
// ---------------------------------------------------------------------------
#define BKP 72   // padded LDS row (elements) to avoid bank conflicts

__global__ __launch_bounds__(256) void gemm_bf16_kernel(const unsigned short* __restrict__ A,
                                                        const unsigned short* __restrict__ W,
                                                        float* __restrict__ Cout, int act) {
    __shared__ __align__(16) unsigned short As[2][128 * BKP];
    __shared__ __align__(16) unsigned short Bs[2][64 * BKP];

    const int tid   = threadIdx.x;
    const int lane  = tid & 31;
    const int wave  = tid >> 5;
    const int wm    = wave & 3;        // wave row  (0..3) -> 32 rows each
    const int wn    = wave >> 2;       // wave col  (0..1) -> 32 cols each
    const int lrow  = lane & 15;
    const int lhalf = lane >> 4;       // 0 or 1

    const size_t aRowBase = (size_t)blockIdx.x * 128;
    const size_t bRowBase = (size_t)blockIdx.y * 64;

    v8f acc[2][2] = {};

    // Per-thread staging coordinates (chunk = 16B = 8 bf16).
    const int ra0 = (tid >> 3);            // A rows for i=0..3: ra0 + i*32
    const int cc8 = (tid & 7) * 8;         // element column within 64-wide tile

    auto stage_async = [&](int buf, int kt) {
        const int kcol = kt * 64 + cc8;
        #pragma unroll
        for (int i = 0; i < 4; ++i) {
            int r = ra0 + i * 32;
            async_b128((unsigned)(size_t)&As[buf][r * BKP + cc8],
                       A + (aRowBase + r) * Cx + kcol);
        }
        #pragma unroll
        for (int i = 0; i < 2; ++i) {
            int r = ra0 + i * 32;
            async_b128((unsigned)(size_t)&Bs[buf][r * BKP + cc8],
                       W + (bRowBase + r) * Cx + kcol);
        }
    };

    // prologue: stage tile 0
    stage_async(0, 0);
    wait_async0();
    __syncthreads();

    for (int kt = 0; kt < Cx / 64; ++kt) {
        const int cur = kt & 1;
        if (kt + 1 < Cx / 64) stage_async(cur ^ 1, kt + 1);   // overlap with compute

        #pragma unroll
        for (int kk = 0; kk < 64; kk += 32) {
            union Frag { v16bf v; uint4 q[2]; } a[2], bw[2];
            // A frag (16-bit 16x32): lanes 0-15 K=0..7/16..23, lanes 16-31 K=8..15/24..31
            #pragma unroll
            for (int mi = 0; mi < 2; ++mi) {
                const unsigned short* ap =
                    &As[cur][(wm * 32 + mi * 16 + lrow) * BKP + kk + lhalf * 8];
                a[mi].q[0] = *(const uint4*)ap;
                a[mi].q[1] = *(const uint4*)(ap + 16);
            }
            // B frag (16-bit 32x16): N = lane&15, lanes 0-15 K=0..7/16..23, lanes 16-31 K=8..15/24..31
            #pragma unroll
            for (int ni = 0; ni < 2; ++ni) {
                const unsigned short* bp =
                    &Bs[cur][(wn * 32 + ni * 16 + lrow) * BKP + kk + lhalf * 8];
                bw[ni].q[0] = *(const uint4*)bp;
                bw[ni].q[1] = *(const uint4*)(bp + 16);
            }
            #pragma unroll
            for (int mi = 0; mi < 2; ++mi)
                #pragma unroll
                for (int ni = 0; ni < 2; ++ni)
                    acc[mi][ni] = __builtin_amdgcn_wmma_f32_16x16x32_bf16(
                        false, a[mi].v, false, bw[ni].v,
                        (short)0, acc[mi][ni], false, false);
        }

        wait_async0();          // next tile landed in LDS (this wave's copies)
        __syncthreads();        // collectively: all waves' copies + reads done
    }

    // ---- epilogue: C layout -> VGPR i holds (row = i + (lane>>4)*8, col = lane&15)
    #pragma unroll
    for (int mi = 0; mi < 2; ++mi) {
        #pragma unroll
        for (int ni = 0; ni < 2; ++ni) {
            size_t rowBase = aRowBase + wm * 32 + mi * 16 + lhalf * 8;
            size_t col     = bRowBase + wn * 32 + ni * 16 + lrow;
            #pragma unroll
            for (int i = 0; i < 8; ++i) {
                float val = acc[mi][ni][i];
                if (act == 1) val = 1.0f / (1.0f + __expf(-val));
                Cout[(rowBase + i) * Cx + col] = val;
            }
        }
    }
}

// ---------------------------------------------------------------------------
// WKV chunked parallel scan (log-semiring affine recurrence).
// Phase 1: per-(b,c,chunk) summary (M, Sv, S1), independent of incoming state
// ---------------------------------------------------------------------------
__global__ void wkv_phase1_kernel(const float* __restrict__ k, const float* __restrict__ v,
                                  const float* __restrict__ sdecay,
                                  float* __restrict__ cM, float* __restrict__ cA,
                                  float* __restrict__ cB) {
    int t     = blockIdx.x * 256 + threadIdx.x;     // B*C*NCH = 65536
    int c     = t & (Cx - 1);
    int rest  = t >> 9;
    int chunk = rest & (NCH - 1);
    int b     = rest >> 4;

    float w = -__expf(sdecay[c] * (1.0f / (float)Nx));
    size_t base = (((size_t)b * Nx + chunk * CT) << 9) + c;

    float la = 0.0f, lb = 0.0f, lp = -1e38f;
    for (int j = 0; j < CT; ++j) {
        float kk = k[base + (size_t)j * Cx];
        float vv = v[base + (size_t)j * Cx];
        float ww2 = lp + w;
        float q2  = fmaxf(ww2, kk);
        float f1  = __expf(ww2 - q2);
        float f2  = __expf(kk - q2);
        la = f1 * la + f2 * vv;
        lb = f1 * lb + f2;
        lp = q2;
    }
    int idx = (b * NCH + chunk) * Cx + c;
    cM[idx] = lp; cA[idx] = la; cB[idx] = lb;
}

// Phase 2: per-(b,c) sequential combine over 16 chunks -> incoming states
__global__ void wkv_phase2_kernel(const float* __restrict__ sdecay,
                                  const float* __restrict__ cM, const float* __restrict__ cA,
                                  const float* __restrict__ cB,
                                  float* __restrict__ iP, float* __restrict__ iA,
                                  float* __restrict__ iB) {
    int t = blockIdx.x * 256 + threadIdx.x;         // B*C = 4096
    int c = t & (Cx - 1);
    int b = t >> 9;
    float w  = -__expf(sdecay[c] * (1.0f / (float)Nx));
    float Tw = (float)CT * w;

    float a = 0.0f, bb = 0.0f, p = -1e38f;
    for (int i = 0; i < NCH; ++i) {
        int idx = (b * NCH + i) * Cx + c;
        iP[idx] = p; iA[idx] = a; iB[idx] = bb;
        float M  = cM[idx], Sv = cA[idx], S1 = cB[idx];
        float pT = p + Tw;
        float pn = fmaxf(pT, M);
        float f  = __expf(pT - pn);
        float g  = __expf(M  - pn);
        a  = f * a  + g * Sv;
        bb = f * bb + g * S1;
        p  = pn;
    }
}

// Phase 3: replay each chunk from its incoming state, emit y = sigmoid(r)*wkv (bf16)
__global__ void wkv_phase3_kernel(const float* __restrict__ k, const float* __restrict__ v,
                                  const float* __restrict__ sr,
                                  const float* __restrict__ sdecay, const float* __restrict__ sfirst,
                                  const float* __restrict__ iP, const float* __restrict__ iA,
                                  const float* __restrict__ iB,
                                  unsigned short* __restrict__ y) {
    int t     = blockIdx.x * 256 + threadIdx.x;     // B*C*NCH
    int c     = t & (Cx - 1);
    int rest  = t >> 9;
    int chunk = rest & (NCH - 1);
    int b     = rest >> 4;

    float w = -__expf(sdecay[c] * (1.0f / (float)Nx));
    float u = sfirst[c] * (1.0f / (float)Nx);

    int idx = (b * NCH + chunk) * Cx + c;
    float a = iA[idx], bb = iB[idx], p = iP[idx];
    size_t base = (((size_t)b * Nx + chunk * CT) << 9) + c;

    for (int j = 0; j < CT; ++j) {
        size_t o = base + (size_t)j * Cx;
        float kk = k[o], vv = v[o], srt = sr[o];
        // output
        float ww = u + kk;
        float q  = fmaxf(p, ww);
        float e1 = __expf(p  - q);
        float e2 = __expf(ww - q);
        float out = (e1 * a + e2 * vv) / (e1 * bb + e2);
        y[o] = f2bf(srt * out);
        // state update
        float ww2 = p + w;
        float q2  = fmaxf(ww2, kk);
        float f1  = __expf(ww2 - q2);
        float f2  = __expf(kk  - q2);
        a  = f1 * a  + f2 * vv;
        bb = f1 * bb + f2;
        p  = q2;
    }
}

// ---------------------------------------------------------------------------
// LayerNorm over channel dim, in place on d_out. One wave per row (wave32).
// ---------------------------------------------------------------------------
__global__ void layernorm_kernel(float* __restrict__ x,
                                 const float* __restrict__ lnw, const float* __restrict__ lnb) {
    int row  = blockIdx.x * 8 + (threadIdx.x >> 5);
    int lane = threadIdx.x & 31;
    float* xr = x + (size_t)row * Cx;

    float v[16];
    float s = 0.0f, s2 = 0.0f;
    #pragma unroll
    for (int j = 0; j < 16; ++j) {
        float t = xr[lane + j * 32];
        v[j] = t; s += t; s2 += t * t;
    }
    #pragma unroll
    for (int m = 16; m >= 1; m >>= 1) {
        s  += __shfl_xor(s,  m, 32);
        s2 += __shfl_xor(s2, m, 32);
    }
    float mu  = s * (1.0f / Cx);
    float var = s2 * (1.0f / Cx) - mu * mu;
    float inv = rsqrtf(var + 1e-5f);
    #pragma unroll
    for (int j = 0; j < 16; ++j) {
        int cc = lane + j * 32;
        xr[cc] = (v[j] - mu) * inv * lnw[cc] + lnb[cc];
    }
}

// ---------------------------------------------------------------------------
extern "C" void kernel_launch(void* const* d_in, const int* in_sizes, int n_in,
                              void* d_out, int out_size, void* d_ws, size_t ws_size,
                              hipStream_t stream) {
    const float* tokens = (const float*)d_in[0];
    const float* mix_k  = (const float*)d_in[1];
    const float* mix_v  = (const float*)d_in[2];
    const float* mix_r  = (const float*)d_in[3];
    const float* w_key  = (const float*)d_in[4];
    const float* w_val  = (const float*)d_in[5];
    const float* w_rec  = (const float*)d_in[6];
    const float* w_out  = (const float*)d_in[7];
    const float* sdecay = (const float*)d_in[8];
    const float* sfirst = (const float*)d_in[9];
    const float* ln_w   = (const float*)d_in[10];
    const float* ln_b   = (const float*)d_in[11];
    float* out = (float*)d_out;

    char* ws = (char*)d_ws;
    size_t off = 0;
    auto carve = [&](size_t bytes) -> void* {
        void* p = ws + off;
        off += (bytes + 255) & ~(size_t)255;
        return p;
    };

    const size_t MC = (size_t)Mx * Cx;
    unsigned short* xk  = (unsigned short*)carve(MC * 2);
    unsigned short* xv  = (unsigned short*)carve(MC * 2);
    unsigned short* xr  = (unsigned short*)carve(MC * 2);
    unsigned short* wkb = (unsigned short*)carve((size_t)Cx * Cx * 2);
    unsigned short* wvb = (unsigned short*)carve((size_t)Cx * Cx * 2);
    unsigned short* wrb = (unsigned short*)carve((size_t)Cx * Cx * 2);
    unsigned short* wob = (unsigned short*)carve((size_t)Cx * Cx * 2);
    float* kf  = (float*)carve(MC * 4);
    float* vf  = (float*)carve(MC * 4);
    float* srf = (float*)carve(MC * 4);
    unsigned short* ybf = (unsigned short*)carve(MC * 2);
    const size_t SC = (size_t)Bx * NCH * Cx * 4;
    float* cM = (float*)carve(SC);
    float* cA = (float*)carve(SC);
    float* cB = (float*)carve(SC);
    float* iP = (float*)carve(SC);
    float* iA = (float*)carve(SC);
    float* iB = (float*)carve(SC);
    (void)ws_size; (void)in_sizes; (void)n_in; (void)out_size;

    // 0. weights -> bf16
    convert_weights_kernel<<<(Cx * Cx) / 256, 256, 0, stream>>>(
        w_key, w_val, w_rec, w_out, wkb, wvb, wrb, wob);

    // 1. q_shift + mix -> bf16 activations
    qshift_mix_kernel<<<(unsigned)(MC / 256), 256, 0, stream>>>(
        tokens, mix_k, mix_v, mix_r, xk, xv, xr);

    // 2. three WMMA GEMMs (k, v, sigmoid(r))
    dim3 ggrid(Mx / 128, Cx / 64);
    gemm_bf16_kernel<<<ggrid, 256, 0, stream>>>(xk, wkb, kf, 0);
    gemm_bf16_kernel<<<ggrid, 256, 0, stream>>>(xv, wvb, vf, 0);
    gemm_bf16_kernel<<<ggrid, 256, 0, stream>>>(xr, wrb, srf, 1);

    // 3. WKV chunked scan
    wkv_phase1_kernel<<<(Bx * Cx * NCH) / 256, 256, 0, stream>>>(kf, vf, sdecay, cM, cA, cB);
    wkv_phase2_kernel<<<(Bx * Cx) / 256, 256, 0, stream>>>(sdecay, cM, cA, cB, iP, iA, iB);
    wkv_phase3_kernel<<<(Bx * Cx * NCH) / 256, 256, 0, stream>>>(
        kf, vf, srf, sdecay, sfirst, iP, iA, iB, ybf);

    // 4. output projection (WMMA) into d_out, then in-place LayerNorm
    gemm_bf16_kernel<<<ggrid, 256, 0, stream>>>(ybf, wob, out, 0);
    layernorm_kernel<<<Mx / 8, 256, 0, stream>>>(out, ln_w, ln_b);
}
